// GIN_40475771797954
// MI455X (gfx1250) — matrix operand order repaired and checked by reading, not compiled, not run on hardware
//
#include <hip/hip_runtime.h>
#include <hip/hip_bf16.h>

typedef __attribute__((ext_vector_type(16))) _Float16 v16h;
typedef __attribute__((ext_vector_type(8)))  float    v8f;

#define N_NODES    50000
#define N_EDGES    800000
#define N_GRAPHS   512
#define EMB        100
#define NLAYERS    5
#define ATOM_FEATS 9
#define ATOM_VOCAB 128
#define BOND_FEATS 3
#define BOND_VOCAB 16
#define NTASK      2
#define BN_EPSF    1e-5f

#define D2   (2*EMB)        // 200
#define KP1  128            // padded K for GEMM1
#define NP1  256            // padded N for GEMM1
#define KP2  256            // padded K for GEMM2
#define NP2  128            // padded N for GEMM2
#define MTILES (N_NODES/16) // 3125 exact

// ---------------------------------------------------------------------------
// elementwise / encoder kernels
// ---------------------------------------------------------------------------
__global__ void k_zero(float* p, int n) {
    int i = blockIdx.x * blockDim.x + threadIdx.x;
    if (i < n) p[i] = 0.f;
}

__global__ void k_atom_encode(const int* __restrict__ xa,
                              const float* __restrict__ emb,
                              float* __restrict__ h) {
    int idx = blockIdx.x * blockDim.x + threadIdx.x;
    if (idx >= N_NODES * EMB) return;
    int n = idx / EMB, d = idx % EMB;
    float s = 0.f;
#pragma unroll
    for (int f = 0; f < ATOM_FEATS; ++f) {
        int v = xa[n * ATOM_FEATS + f];
        s += emb[((size_t)f * ATOM_VOCAB + v) * EMB + d];
    }
    h[idx] = s;
}

__global__ void k_hh_init(const float* __restrict__ h,
                          const float* __restrict__ eps, int layer,
                          float* __restrict__ hh) {
    int idx = blockIdx.x * blockDim.x + threadIdx.x;
    if (idx >= N_NODES * EMB) return;
    hh[idx] = (1.f + eps[layer]) * h[idx];
}

// msg = h[src] + bond_embed(edge_attr); hh[dst] += msg  (scatter-add).
// Bond table (19.2 KB) staged in LDS.  One wave per edge, grid-stride with
// 4 edges per wave; next edge's h[src] row is prefetched (4 cachelines,
// one per lane group) while the current edge is processed.
__global__ void k_edge_scatter(const int* __restrict__ eidx,
                               const int* __restrict__ eattr,
                               const float* __restrict__ bond,
                               const float* __restrict__ h,
                               float* __restrict__ hh) {
    __shared__ float sb[BOND_FEATS * BOND_VOCAB * EMB];
    int tid = threadIdx.x;
    for (int i = tid; i < BOND_FEATS * BOND_VOCAB * EMB; i += blockDim.x)
        sb[i] = bond[i];
    __syncthreads();
    const int wave = tid >> 5, lane = tid & 31;
    const int wpb = blockDim.x >> 5;
    const int stride = gridDim.x * wpb;
    for (int e = blockIdx.x * wpb + wave; e < N_EDGES; e += stride) {
        int src = eidx[e];
        int dst = eidx[N_EDGES + e];
        int en = e + stride;                 // prefetch next edge's source row
        if (en < N_EDGES) {
            int nsrc = eidx[en];
            __builtin_prefetch(&h[(size_t)nsrc * EMB + (lane & 3) * 32], 0, 0);
        }
        int a0 = eattr[e * BOND_FEATS + 0];
        int a1 = eattr[e * BOND_FEATS + 1];
        int a2 = eattr[e * BOND_FEATS + 2];
        const float* b0 = &sb[(0 * BOND_VOCAB + a0) * EMB];
        const float* b1 = &sb[(1 * BOND_VOCAB + a1) * EMB];
        const float* b2 = &sb[(2 * BOND_VOCAB + a2) * EMB];
        for (int d = lane; d < EMB; d += 32) {
            float msg = h[(size_t)src * EMB + d] + b0[d] + b1[d] + b2[d];
            atomicAdd(&hh[(size_t)dst * EMB + d], msg);
        }
    }
}

// ---------------------------------------------------------------------------
// WMMA fragment packing
// A-fragment layout (16-bit A, 16x32, ISA 7.12.2): element i of lane L
// (half=L/16) holds K = i + 8*((i>>3)+half).  Inverse: given kk in [0,32):
//   half = (kk>>3)&1 ;  i = kk - 8*(kk>>4) - 8*half ; lane = (m&15)+16*half
// B-fragment layout (32x16): element i of lane L holds K = i + 8*((i>>3)+half), N=L%16
// ---------------------------------------------------------------------------
__device__ __forceinline__ void a_frag_pos(int kk, int mrow, int& lane, int& i) {
    int half = (kk >> 3) & 1;
    i = kk - 8 * (kk >> 4) - 8 * half;
    lane = (mrow & 15) + 16 * half;
}

// pack f32 activations [M, srcCols] -> f16 A fragments [mt][kt][lane][i]
__global__ void k_pack_act(const float* __restrict__ X, int srcCols,
                           _Float16* __restrict__ dst, int KP, int Ktiles) {
    int idx = blockIdx.x * blockDim.x + threadIdx.x;
    if (idx >= N_NODES * KP) return;
    int m = idx / KP, kk = idx % KP;
    float v = (kk < srcCols) ? X[(size_t)m * srcCols + kk] : 0.f;
    int kt = kk >> 5, q = kk & 31, lane, i;
    a_frag_pos(q, m, lane, i);
    int mt = m >> 4;
    dst[(((size_t)mt * Ktiles + kt) * 32 + lane) * 16 + i] = (_Float16)v;
}

// BN + ReLU + f16-convert + A-fragment pack (GEMM1 output -> GEMM2 input)
__global__ void k_bn_relu_pack(const float* __restrict__ X, int ldc, int ncols,
                               const float* __restrict__ csum,
                               const float* __restrict__ csq,
                               const float* __restrict__ gamma,
                               const float* __restrict__ beta,
                               _Float16* __restrict__ dst, int KP, int Ktiles) {
    int idx = blockIdx.x * blockDim.x + threadIdx.x;
    if (idx >= N_NODES * KP) return;
    int m = idx / KP, kk = idx % KP;
    float v = 0.f;
    if (kk < ncols) {
        const float invN = 1.f / (float)N_NODES;
        float mu  = csum[kk] * invN;
        float var = csq[kk] * invN - mu * mu;
        float x = X[(size_t)m * ldc + kk];
        float z = (x - mu) * rsqrtf(var + BN_EPSF) * gamma[kk] + beta[kk];
        v = fmaxf(z, 0.f);
    }
    int kt = kk >> 5, q = kk & 31, lane, i;
    a_frag_pos(q, m, lane, i);
    int mt = m >> 4;
    dst[(((size_t)mt * Ktiles + kt) * 32 + lane) * 16 + i] = (_Float16)v;
}

// pack f32 weights [krows, ncols] -> f16 B fragments [kt][nt][lane][i]
__global__ void k_pack_w(const float* __restrict__ W, int krows, int ncols,
                         _Float16* __restrict__ dst, int Ktiles, int Ntiles) {
    int idx = blockIdx.x * blockDim.x + threadIdx.x;
    if (idx >= Ktiles * Ntiles * 32 * 16) return;
    int i    = idx & 15;
    int lane = (idx >> 4) & 31;
    int rem  = idx >> 9;
    int nt = rem % Ntiles, kt = rem / Ntiles;
    int half = lane >> 4, c = lane & 15;
    int k = kt * 32 + i + 8 * half + 8 * (i >> 3);
    int n = nt * 16 + c;
    float v = (k < krows && n < ncols) ? W[(size_t)k * ncols + n] : 0.f;
    dst[idx] = (_Float16)v;
}

// ---------------------------------------------------------------------------
// WMMA GEMM: C[M, ldc] = Apack x Bpack + bias.  4 waves/block, each wave owns
// a 16x64 tile (4 f32 accumulators).  Compile-time Ktiles/Ntiles -> full
// unroll; all A fragments preloaded into registers so the scheduler can batch
// global loads instead of load->wait->wmma serialization.
// ---------------------------------------------------------------------------
template <int KT, int NTILES>
__global__ void k_wmma_gemm(const _Float16* __restrict__ Ap,
                            const _Float16* __restrict__ Bp,
                            const float* __restrict__ bias, int biasValid,
                            float* __restrict__ C, int Mtiles, int ldc) {
    int wave = threadIdx.x >> 5, lane = threadIdx.x & 31;
    int mt  = blockIdx.x * 4 + wave;
    int nt0 = blockIdx.y * 4;
    if (mt >= Mtiles) return;

    const v16h* A = (const v16h*)Ap + (size_t)mt * KT * 32 + lane;
    const v16h* B = (const v16h*)Bp + lane;

    v16h a[KT];
#pragma unroll
    for (int kt = 0; kt < KT; ++kt) a[kt] = A[kt * 32];

    v8f acc[4] = {};
#pragma unroll
    for (int kt = 0; kt < KT; ++kt) {
#pragma unroll
        for (int j = 0; j < 4; ++j) {
            v16h b = B[((size_t)kt * NTILES + nt0 + j) * 32];
            acc[j] = __builtin_amdgcn_wmma_f32_16x16x32_f16(
                false, a[kt], false, b, (short)0, acc[j], false, false);
        }
    }

    // 16x16 f32 C/D layout: VGPR j -> row m0 + j + 8*half, col = lane%16
    int half = lane >> 4, c = lane & 15;
    int m0 = mt * 16;
#pragma unroll
    for (int j = 0; j < 4; ++j) {
        int n = (nt0 + j) * 16 + c;
        float bv = (n < biasValid) ? bias[n] : 0.f;
#pragma unroll
        for (int r = 0; r < 8; ++r)
            C[(size_t)(m0 + r + 8 * half) * ldc + n] = acc[j][r] + bv;
    }
}

// ---------------------------------------------------------------------------
// BatchNorm statistics (column sum / sum-of-squares via atomics)
// ---------------------------------------------------------------------------
__global__ void k_bn_stats(const float* __restrict__ X, int ldc, int ncols,
                           float* __restrict__ sum, float* __restrict__ sq) {
    int c = threadIdx.x;
    if (c >= ncols) return;
    int r0 = blockIdx.x * 64;
    int r1 = r0 + 64; if (r1 > N_NODES) r1 = N_NODES;
    float s = 0.f, s2 = 0.f;
    for (int r = r0; r < r1; ++r) {
        float x = X[(size_t)r * ldc + c];
        s += x; s2 += x * x;
    }
    atomicAdd(&sum[c], s);
    atomicAdd(&sq[c], s2);
}

// BN + ReLU into dense f32 h [N, EMB]  (outer BatchNorm of each layer)
__global__ void k_bn_relu_out(const float* __restrict__ X, int ldc,
                              const float* __restrict__ csum,
                              const float* __restrict__ csq,
                              const float* __restrict__ gamma,
                              const float* __restrict__ beta,
                              float* __restrict__ h) {
    int idx = blockIdx.x * blockDim.x + threadIdx.x;
    if (idx >= N_NODES * EMB) return;
    int m = idx / EMB, d = idx % EMB;
    const float invN = 1.f / (float)N_NODES;
    float mu  = csum[d] * invN;
    float var = csq[d] * invN - mu * mu;
    float x = X[(size_t)m * ldc + d];
    float z = (x - mu) * rsqrtf(var + BN_EPSF) * gamma[d] + beta[d];
    h[idx] = fmaxf(z, 0.f);
}

// ---------------------------------------------------------------------------
// global mean pool + linear head
// ---------------------------------------------------------------------------
__global__ void k_pool(const float* __restrict__ h, const int* __restrict__ batch,
                       float* __restrict__ gsum, float* __restrict__ gcnt) {
    int wave = threadIdx.x >> 5, lane = threadIdx.x & 31;
    int n = blockIdx.x * (blockDim.x >> 5) + wave;
    if (n >= N_NODES) return;
    int g = batch[n];
    for (int d = lane; d < EMB; d += 32)
        atomicAdd(&gsum[(size_t)g * EMB + d], h[(size_t)n * EMB + d]);
    if (lane == 0) atomicAdd(&gcnt[g], 1.f);
}

__global__ void k_head(const float* __restrict__ gsum,
                       const float* __restrict__ gcnt,
                       const float* __restrict__ Wp,
                       const float* __restrict__ bp,
                       float* __restrict__ out) {
    int idx = blockIdx.x * blockDim.x + threadIdx.x;
    if (idx >= N_GRAPHS * NTASK) return;
    int g = idx / NTASK, t = idx % NTASK;
    float inv = 1.f / fmaxf(gcnt[g], 1.f);
    float s = bp[t];
    for (int d = 0; d < EMB; ++d)
        s += gsum[(size_t)g * EMB + d] * inv * Wp[d * NTASK + t];
    out[idx] = s;
}

// ---------------------------------------------------------------------------
// host launcher
// ---------------------------------------------------------------------------
extern "C" void kernel_launch(void* const* d_in, const int* in_sizes, int n_in,
                              void* d_out, int out_size, void* d_ws, size_t ws_size,
                              hipStream_t stream) {
    const int*   x_atom    = (const int*)d_in[0];
    const int*   edge_idx  = (const int*)d_in[1];
    const int*   edge_attr = (const int*)d_in[2];
    const int*   batch     = (const int*)d_in[3];
    const float* atom_emb  = (const float*)d_in[4];
    const float* bond_emb  = (const float*)d_in[5];
    const float* eps       = (const float*)d_in[6];
    const float* W1  = (const float*)d_in[7];
    const float* b1  = (const float*)d_in[8];
    const float* g1  = (const float*)d_in[9];
    const float* be1 = (const float*)d_in[10];
    const float* W2  = (const float*)d_in[11];
    const float* b2  = (const float*)d_in[12];
    const float* go  = (const float*)d_in[13];
    const float* bo  = (const float*)d_in[14];
    const float* Wp  = (const float*)d_in[15];
    const float* bp  = (const float*)d_in[16];
    float* out = (float*)d_out;

    // workspace carve-up (~155 MB total; whole working set is L2-resident)
    size_t off = 0;
    auto carve = [&](size_t bytes) -> char* {
        char* p = (char*)d_ws + off;
        off += (bytes + 255) & ~(size_t)255;
        return p;
    };
    float*    h    = (float*)carve((size_t)N_NODES * EMB * 4);
    float*    hh   = (float*)carve((size_t)N_NODES * EMB * 4);
    float*    zpre = (float*)carve((size_t)N_NODES * NP1 * 4);
    float*    hpre = (float*)carve((size_t)N_NODES * NP2 * 4);
    _Float16* Ap1  = (_Float16*)carve((size_t)MTILES * (KP1/32) * 512 * 2);
    _Float16* Ap2  = (_Float16*)carve((size_t)MTILES * (KP2/32) * 512 * 2);
    _Float16* Bp1  = (_Float16*)carve((size_t)(KP1/32) * (NP1/16) * 512 * 2);
    _Float16* Bp2  = (_Float16*)carve((size_t)(KP2/32) * (NP2/16) * 512 * 2);
    float*    csum = (float*)carve(256 * 4);
    float*    csq  = (float*)carve(256 * 4);
    float*    gsum = (float*)carve((size_t)N_GRAPHS * EMB * 4);
    float*    gcnt = (float*)carve((size_t)N_GRAPHS * 4);

    const int TB = 256;
    const int NE_GRID = (N_NODES * EMB + TB - 1) / TB;

    // atom encoder -> h
    k_atom_encode<<<NE_GRID, TB, 0, stream>>>(x_atom, atom_emb, h);

    for (int l = 0; l < NLAYERS; ++l) {
        // weights -> f16 B fragments
        k_pack_w<<<((KP1/32)*(NP1/16)*512 + TB-1)/TB, TB, 0, stream>>>(
            W1 + (size_t)l * EMB * D2, EMB, D2, Bp1, KP1/32, NP1/16);
        k_pack_w<<<((KP2/32)*(NP2/16)*512 + TB-1)/TB, TB, 0, stream>>>(
            W2 + (size_t)l * D2 * EMB, D2, EMB, Bp2, KP2/32, NP2/16);

        // hh = (1+eps)*h ; then scatter-add edge messages (4 edges/wave)
        k_hh_init<<<NE_GRID, TB, 0, stream>>>(h, eps, l, hh);
        k_edge_scatter<<<N_EDGES / 8 / 4, TB, 0, stream>>>(edge_idx, edge_attr,
                                                           bond_emb, h, hh);

        // hh -> f16 A fragments (K padded 100->128)
        k_pack_act<<<(N_NODES*KP1 + TB-1)/TB, TB, 0, stream>>>(hh, EMB, Ap1,
                                                               KP1, KP1/32);
        // GEMM1: zpre = hh @ W1 + b1     [50000 x 256], cols >=200 are 0
        dim3 gg1((MTILES + 3) / 4, NP1 / 64);
        k_wmma_gemm<KP1/32, NP1/16><<<gg1, 128, 0, stream>>>(
            Ap1, Bp1, b1 + (size_t)l * D2, D2, zpre, MTILES, NP1);

        // BN stats over zpre (200 valid cols)
        k_zero<<<1, TB, 0, stream>>>(csum, 256);
        k_zero<<<1, TB, 0, stream>>>(csq, 256);
        k_bn_stats<<<(N_NODES + 63) / 64, TB, 0, stream>>>(zpre, NP1, D2, csum, csq);

        // BN + ReLU + pack -> A fragments for GEMM2 (K padded 200->256)
        k_bn_relu_pack<<<(N_NODES*KP2 + TB-1)/TB, TB, 0, stream>>>(
            zpre, NP1, D2, csum, csq, g1 + (size_t)l * D2, be1 + (size_t)l * D2,
            Ap2, KP2, KP2/32);

        // GEMM2: hpre = z @ W2 + b2      [50000 x 128], cols >=100 are 0
        dim3 gg2((MTILES + 3) / 4, NP2 / 64);
        k_wmma_gemm<KP2/32, NP2/16><<<gg2, 128, 0, stream>>>(
            Ap2, Bp2, b2 + (size_t)l * EMB, EMB, hpre, MTILES, NP2);

        // outer BN stats + BN+ReLU -> h
        k_zero<<<1, TB, 0, stream>>>(csum, 256);
        k_zero<<<1, TB, 0, stream>>>(csq, 256);
        k_bn_stats<<<(N_NODES + 63) / 64, TB, 0, stream>>>(hpre, NP2, EMB, csum, csq);
        k_bn_relu_out<<<NE_GRID, TB, 0, stream>>>(hpre, NP2, csum, csq,
                                                  go + (size_t)l * EMB,
                                                  bo + (size_t)l * EMB, h);
    }

    // global mean pool + head
    k_zero<<<(N_GRAPHS*EMB + TB-1)/TB, TB, 0, stream>>>(gsum, N_GRAPHS * EMB);
    k_zero<<<(N_GRAPHS + TB-1)/TB, TB, 0, stream>>>(gcnt, N_GRAPHS);
    k_pool<<<N_NODES/8, TB, 0, stream>>>(h, batch, gsum, gcnt);
    k_head<<<(N_GRAPHS*NTASK + TB-1)/TB, TB, 0, stream>>>(gsum, gcnt, Wp, bp, out);
}
